// GraphTransformerMatching_22393959482126
// MI455X (gfx1250) — matrix operand order repaired and checked by reading, not compiled, not run on hardware
//
#include <hip/hip_runtime.h>

#define NB   8
#define NSN  512
#define MTOT 4096        // B*NS = B*NT
#define NE   32768       // edges
#define EMB  128
#define ORI_ 1024
#define KK_  25
#define CIN  256         // 2*EMB
#define YW   3200        // KK_*EMB
#define FIN  1408        // ORI + 3*EMB
#define NTW  4           // 16-wide n-tiles per wave (64 columns)

typedef _Float16 h16;
typedef __attribute__((ext_vector_type(16))) _Float16 v16h;
typedef __attribute__((ext_vector_type(8)))  _Float16 v8h;
typedef __attribute__((ext_vector_type(8)))  float    v8f;

// ---------------------------------------------------------------------------
// WMMA f16 GEMM, wave computes a 16x64 tile (4 accumulators), f32 accumulate.
//   C[M,N] = act( gemmScale*(A@B) + bias[N] + addend[M,N]*rowscale[M] )
// A : f16 row-major [M,K], row stride lda, batch stride aBatch
//     (batch b = row / rowsPerBatch)
// BT: f16 TRANSPOSED [N,K], row stride ldbt, batch stride bBatch
//     -> both fragments load as K-contiguous 16B vectors (global_load_b128)
// Optional stores: f32 C (ldc), f16 Ch (ldch), per-batch transposed f16
// ChT[b][col][localRow] (ldcht, cBatchT).
// Requires: M %16==0, N %64==0, K %32==0, rowsPerBatch %16==0,
//           A/BT rows 16-byte aligned.
// ---------------------------------------------------------------------------
__global__ void __launch_bounds__(256)
wmma_gemm_f16(const h16* __restrict__ A, int lda, long aBatch,
              const h16* __restrict__ BT, int ldbt, long bBatch,
              int M, int N, int Kp, int rowsPerBatch,
              const float* __restrict__ bias,
              const float* __restrict__ addend, int ldadd,
              const float* __restrict__ rowscale,
              float gemmScale, int doTanh,
              float* __restrict__ C, int ldc,
              h16* __restrict__ Ch, int ldch,
              h16* __restrict__ ChT, int ldcht, long cBatchT)
{
    int tilesNW = N >> 6;                    // groups of 4 n-tiles
    int tiles   = (M >> 4) * tilesNW;
    int wave    = (int)((blockIdx.x * blockDim.x + threadIdx.x) >> 5);
    if (wave >= tiles) return;               // whole wave exits together
    int tm   = (wave / tilesNW) << 4;
    int tn0  = (wave % tilesNW) << 6;
    int lane = threadIdx.x & 31;
    int hf   = lane >> 4;                    // half-wave select
    int l16  = lane & 15;

    int b      = tm / rowsPerBatch;
    int rowA   = tm + l16;                   // A-fragment row: M = lane%16
    int localA = rowA - b * rowsPerBatch;

    const h16* Ar = A  + (long)b * aBatch + (long)localA * lda;
    const h16* Bb = BT + (long)b * bBatch + (long)(tn0 + l16) * ldbt;

    v8f acc[NTW];
#pragma unroll
    for (int tI = 0; tI < NTW; ++tI)
        acc[tI] = (v8f){0.f,0.f,0.f,0.f,0.f,0.f,0.f,0.f};

    for (int k0 = 0; k0 < Kp; k0 += 32) {
        // A 16-bit 16x32 fragment: K groups {8*hf+0..7, 16+8*hf+0..7}
        v8h a0 = *(const v8h*)(Ar + k0 + 8 * hf);
        v8h a1 = *(const v8h*)(Ar + k0 + 16 + 8 * hf);
        v16h af = __builtin_shufflevector(a0, a1,
                    0,1,2,3,4,5,6,7,8,9,10,11,12,13,14,15);
#pragma unroll
        for (int tI = 0; tI < NTW; ++tI) {
            // B 16-bit 32x16 fragment: N = lane%16, K groups {8*hf+0..7, 16+8*hf+0..7}
            const h16* Bp = Bb + (long)tI * 16 * ldbt + k0;
            v8h b0 = *(const v8h*)(Bp + 8 * hf);
            v8h b1 = *(const v8h*)(Bp + 16 + 8 * hf);
            v16h bf = __builtin_shufflevector(b0, b1,
                        0,1,2,3,4,5,6,7,8,9,10,11,12,13,14,15);
            acc[tI] = __builtin_amdgcn_wmma_f32_16x16x32_f16(
                          false, af, false, bf, (short)0, acc[tI], false, false);
        }
    }

#pragma unroll
    for (int tI = 0; tI < NTW; ++tI) {
        int col = tn0 + tI * 16 + l16;
#pragma unroll
        for (int r = 0; r < 8; ++r) {
            int row = tm + r + 8 * hf;       // D layout: M = r + 8*half
            float v = gemmScale * acc[tI][r];
            if (bias)   v += bias[col];
            if (addend) {
                float sc = rowscale ? rowscale[row] : 1.0f;
                v += addend[(long)row * ldadd + col] * sc;
            }
            if (doTanh) v = tanhf(v);
            if (C)   C  [(long)row * ldc  + col] = v;
            if (Ch)  Ch [(long)row * ldch + col] = (h16)v;
            if (ChT) ChT[(long)b * cBatchT + (long)col * ldcht
                         + (row - b * rowsPerBatch)] = (h16)v;
        }
    }
}

// ---------------------------------------------------------------------------
// Sinusoidal embeddings
// ---------------------------------------------------------------------------
__global__ void time_emb_k(const float* __restrict__ t, float* __restrict__ temb)
{
    int i = blockIdx.x * blockDim.x + threadIdx.x;
    if (i >= NB * 20) return;
    int b = i / 20, c = i % 20, j = c % 10;
    float f = __expf(-__logf(10000.f) / 9.f * (float)j);
    float a = t[b] * f;
    temb[i] = (c < 10) ? __sinf(a) : __cosf(a);
}

// one wave per node: reduce a row (transposed=0) or column (transposed=1) of Xt,
// then emit the 20-dim sinusoidal embedding of sum/10.
__global__ void node_emb_k(const float* __restrict__ Xt, float* __restrict__ num,
                           int transposed)
{
    int wave = (int)((blockIdx.x * blockDim.x + threadIdx.x) >> 5);
    int lane = threadIdx.x & 31;
    if (wave >= MTOT) return;
    int b = wave >> 9, i = wave & 511;
    const float* base = Xt + (long)b * NSN * NSN + (transposed ? i : (long)i * NSN);
    long stride = transposed ? NSN : 1;
    float s = 0.f;
    for (int j = lane; j < NSN; j += 32) s += base[(long)j * stride];
    for (int off = 16; off; off >>= 1) s += __shfl_down(s, off, 32);
    s = __shfl(s, 0, 32);
    if (lane < 20) {
        int j = lane % 10;
        float f = __expf(-__logf(10000.f) / 9.f * (float)j);
        float a = (s * 0.1f) * f;
        num[(long)wave * 20 + lane] = (lane < 10) ? __sinf(a) : __cosf(a);
    }
}

// ---------------------------------------------------------------------------
// B-spline (degree 1, K=5, dim=2) taps: 4 (index, weight) pairs per edge
// ---------------------------------------------------------------------------
__global__ void basis_k(const float* __restrict__ ea,
                        int* __restrict__ tidx, float* __restrict__ tw)
{
    int e = blockIdx.x * blockDim.x + threadIdx.x;
    if (e >= NE) return;
    float v0 = ea[2 * e] * 4.f, v1 = ea[2 * e + 1] * 4.f;
    float l0 = floorf(v0), l1 = floorf(v1);
    float f0 = v0 - l0,   f1 = v1 - l1;
    int i0 = (int)l0, i1 = (int)l1;
#pragma unroll
    for (int s1 = 0; s1 < 2; ++s1)
#pragma unroll
        for (int s0 = 0; s0 < 2; ++s0) {
            int a0 = min(max(i0 + s0, 0), 4);
            int a1 = min(max(i1 + s1, 0), 4);
            float w = (s0 ? f0 : 1.f - f0) * (s1 ? f1 : 1.f - f1);
            int tp = s1 * 2 + s0;
            tidx[e * 4 + tp] = a0 + 5 * a1;
            tw  [e * 4 + tp] = w;
        }
}

__global__ void deg_count_k(const int* __restrict__ ei, int* __restrict__ deg)
{
    int e = blockIdx.x * blockDim.x + threadIdx.x;
    if (e >= NE) return;
    atomicAdd(&deg[ei[NE + e]], 1);
}

__global__ void inv_deg_k(const int* __restrict__ deg, float* __restrict__ inv)
{
    int i = blockIdx.x * blockDim.x + threadIdx.x;
    if (i >= MTOT) return;
    int d = deg[i];
    inv[i] = 1.f / (float)(d > 1 ? d : 1);
}

// per (edge, channel): gather 4 weighted rows of Y[src], scatter-add to agg[dst]
__global__ void edge_agg_k(const int* __restrict__ ei,
                           const int* __restrict__ tidx,
                           const float* __restrict__ tw,
                           const float* __restrict__ Y,
                           float* __restrict__ agg)
{
    int gid = blockIdx.x * blockDim.x + threadIdx.x;
    int e = gid >> 7, c = gid & 127;
    if (e >= NE) return;
    int src = ei[e], dst = ei[NE + e];
    const float* Ys = Y + (long)src * YW;
    float a = 0.f;
#pragma unroll
    for (int tp = 0; tp < 4; ++tp)
        a += tw[e * 4 + tp] * Ys[tidx[e * 4 + tp] * EMB + c];
    __hip_atomic_fetch_add(&agg[(long)dst * EMB + c], a,
                           __ATOMIC_RELAXED, __HIP_MEMORY_SCOPE_AGENT);
}

// ---------------------------------------------------------------------------
// Converters / packers (f32 -> f16)
// ---------------------------------------------------------------------------
// in[Kin,N] f32 -> outT[N,Kpad] f16 (transposed, zero-padded K)
__global__ void cvt_pad_T_k(const float* __restrict__ in, h16* __restrict__ outT,
                            int Kin, int Kpad, int N)
{
    long tid = (long)blockIdx.x * blockDim.x + threadIdx.x;
    long total = (long)N * Kpad;
    if (tid >= total) return;
    long n = tid / Kpad, r = tid % Kpad;
    outT[tid] = (r < Kin) ? (h16)in[r * N + n] : (h16)0.f;
}

__global__ void cvt_xt_k(const float* __restrict__ Xt,
                         h16* __restrict__ Xh, h16* __restrict__ XTh)
{
    long tid = (long)blockIdx.x * blockDim.x + threadIdx.x;
    if (tid >= (long)NB * NSN * NSN) return;
    float v = Xt[tid];
    Xh[tid] = (h16)v;
    long b = tid >> 18, rem = tid & 262143, s = rem >> 9, t = rem & 511;
    XTh[(b << 18) + (t << 9) + s] = (h16)v;
}

// conv_w[k][c][n] -> WyT[(k*128+n)][c]  (transposed B for Y = cat @ Wy)
__global__ void cvt_wyT_k(const float* __restrict__ w, h16* __restrict__ wyT)
{
    long tid = (long)blockIdx.x * blockDim.x + threadIdx.x;
    if (tid >= (long)KK_ * CIN * EMB) return;
    long k = tid / (CIN * EMB), rem = tid % (CIN * EMB);
    long c = rem / EMB, n = rem % EMB;
    wyT[((long)k * EMB + n) * CIN + c] = (h16)w[tid];
}

// A_lin[row][0:W1]=prev, [W1:W1+20]=num, [W1+20:W1+40]=time, rest zero
__global__ void pack_lin_k(const float* __restrict__ prev, int W1,
                           const float* __restrict__ num,
                           const float* __restrict__ temb,
                           h16* __restrict__ out, int Kpad)
{
    long tid = (long)blockIdx.x * blockDim.x + threadIdx.x;
    long total = (long)MTOT * Kpad;
    if (tid >= total) return;
    long row = tid / Kpad, c = tid % Kpad;
    float v;
    if (c < W1)            v = prev[row * W1 + c];
    else if (c < W1 + 20)  v = num [row * 20 + (c - W1)];
    else if (c < W1 + 40)  v = temb[(row >> 9) * 20 + (c - W1 - 20)];
    else                   v = 0.f;
    out[tid] = (h16)v;
}

// A_fin[row] = [x0(1024) | l0(128) | l1(128) | l2(128)]
__global__ void pack_fin_k(const float* __restrict__ x0,
                           const float* __restrict__ l0,
                           const float* __restrict__ l1,
                           const float* __restrict__ l2,
                           h16* __restrict__ out)
{
    long tid = (long)blockIdx.x * blockDim.x + threadIdx.x;
    if (tid >= (long)MTOT * FIN) return;
    long row = tid / FIN, c = tid % FIN;
    float v;
    if (c < ORI_) v = x0[row * ORI_ + c];
    else {
        int li = (int)((c - ORI_) >> 7), cc = (int)((c - ORI_) & 127);
        const float* p = (li == 0) ? l0 : (li == 1) ? l1 : l2;
        v = p[row * EMB + cc];
    }
    out[tid] = (h16)v;
}

// ---------------------------------------------------------------------------
// Row softmax over 512-wide rows, in place (mask is all-true)
// ---------------------------------------------------------------------------
__global__ void softmax512_k(float* __restrict__ out)
{
    __shared__ float red[256];
    float* p = out + (long)blockIdx.x * NSN;
    int t = threadIdx.x;
    float a = p[t], b = p[t + 256];
    red[t] = fmaxf(a, b);
    __syncthreads();
    for (int s = 128; s; s >>= 1) {
        if (t < s) red[t] = fmaxf(red[t], red[t + s]);
        __syncthreads();
    }
    float m = red[0];
    __syncthreads();
    float ea = __expf(a - m), eb = __expf(b - m);
    red[t] = ea + eb;
    __syncthreads();
    for (int s = 128; s; s >>= 1) {
        if (t < s) red[t] += red[t + s];
        __syncthreads();
    }
    float inv = 1.f / red[0];
    p[t] = ea * inv;
    p[t + 256] = eb * inv;
}

// ---------------------------------------------------------------------------
// Host side
// ---------------------------------------------------------------------------
static inline void gemm(hipStream_t st, const h16* A, int lda, long aB,
                        const h16* BT, int ldbt, long bB,
                        int M, int N, int K, int rpb,
                        const float* bias, const float* add, int ldadd,
                        const float* rs, float gs, int th,
                        float* C, int ldc, h16* Ch, int ldch,
                        h16* ChT = nullptr, int ldcht = 0, long cBT = 0)
{
    long tiles = (long)(M / 16) * (N / 64);
    unsigned blocks = (unsigned)((tiles * 32 + 255) / 256);
    wmma_gemm_f16<<<blocks, 256, 0, st>>>(A, lda, aB, BT, ldbt, bB, M, N, K, rpb,
                                          bias, add, ldadd, rs, gs, th,
                                          C, ldc, Ch, ldch, ChT, ldcht, cBT);
}

#define GRID1(n, b) dim3((unsigned)(((long)(n) + (b) - 1) / (b)))

extern "C" void kernel_launch(void* const* d_in, const int* in_sizes, int n_in,
                              void* d_out, int out_size, void* d_ws, size_t ws_size,
                              hipStream_t stream)
{
    const float* t_in      = (const float*)d_in[0];
    const float* Xt        = (const float*)d_in[1];
    const float* x_s       = (const float*)d_in[2];
    const float* x_t       = (const float*)d_in[3];
    const float* ea_s      = (const float*)d_in[4];
    const float* ea_t      = (const float*)d_in[5];
    const float* lin0_w    = (const float*)d_in[6];
    const float* lin0_b    = (const float*)d_in[7];
    const float* lin_w     = (const float*)d_in[8];
    const float* lin_b     = (const float*)d_in[9];
    const float* conv_w    = (const float*)d_in[10];
    const float* conv_root = (const float*)d_in[11];
    const float* conv_bias = (const float*)d_in[12];
    const float* final_w   = (const float*)d_in[13];
    const float* final_b   = (const float*)d_in[14];
    const int*   ei_s      = (const int*)d_in[15];
    const int*   ei_t      = (const int*)d_in[16];
    float* out = (float*)d_out;

    // ---- workspace carve ----
    char* wp = (char*)d_ws;
    auto alloc = [&](size_t b) -> void* {
        void* p = (void*)wp;
        wp += (b + 255) & ~(size_t)255;
        return p;
    };
    float* temb   = (float*)alloc(NB * 20 * 4);
    float* num_s  = (float*)alloc((long)MTOT * 20 * 4);
    float* num_t  = (float*)alloc((long)MTOT * 20 * 4);
    int*   tidx_s = (int*)  alloc((long)NE * 4 * 4);
    float* tw_s   = (float*)alloc((long)NE * 4 * 4);
    int*   tidx_t = (int*)  alloc((long)NE * 4 * 4);
    float* tw_t   = (float*)alloc((long)NE * 4 * 4);
    int*   deg_s  = (int*)  alloc(MTOT * 4);
    int*   deg_t  = (int*)  alloc(MTOT * 4);
    float* invd_s = (float*)alloc(MTOT * 4);
    float* invd_t = (float*)alloc(MTOT * 4);
    h16*   Xh     = (h16*)  alloc((long)NB * NSN * NSN * 2);
    h16*   XTh    = (h16*)  alloc((long)NB * NSN * NSN * 2);
    h16*   w0hT   = (h16*)  alloc((long)EMB * 1088 * 2);   // [128][1088]
    h16*   wl0hT  = (h16*)  alloc((long)EMB * 192 * 2);    // [128][192]
    h16*   wl1hT  = (h16*)  alloc((long)EMB * 192 * 2);
    h16*   finwhT = (h16*)  alloc((long)EMB * FIN * 2);    // [128][1408]
    h16*   wyhT   = (h16*)  alloc((long)YW * CIN * 2);     // [3200][256]
    h16*   roothT = (h16*)  alloc((long)EMB * CIN * 2);    // [128][256]
    h16*   Als    = (h16*)  alloc((long)MTOT * 1088 * 2);
    h16*   Alt    = (h16*)  alloc((long)MTOT * 1088 * 2);
    float* xin_s  = (float*)alloc((long)MTOT * EMB * 4);
    float* xin_t  = (float*)alloc((long)MTOT * EMB * 4);
    h16*   cat_s  = (h16*)  alloc((long)MTOT * CIN * 2);
    h16*   cat_t  = (h16*)  alloc((long)MTOT * CIN * 2);
    h16*   xpT_s  = (h16*)  alloc((long)MTOT * EMB * 2);   // [8][128][512]
    h16*   xpT_t  = (h16*)  alloc((long)MTOT * EMB * 2);
    float* agg    = (float*)alloc((long)MTOT * EMB * 4);
    float* xls[3], *xlt[3];
    for (int i = 0; i < 3; ++i) xls[i] = (float*)alloc((long)MTOT * EMB * 4);
    for (int i = 0; i < 3; ++i) xlt[i] = (float*)alloc((long)MTOT * EMB * 4);
    float* Y = (float*)alloc((long)MTOT * YW * 4);   // 52.4 MB
    // final-stage buffers overlay the Y region (Y dead after last conv)
    const size_t AFB = (size_t)MTOT * FIN * 2;        // 11.5 MB
    h16* Afs = (h16*)((char*)Y);
    h16* Aft = (h16*)((char*)Y + AFB);
    h16* hsh = (h16*)((char*)Y + 2 * AFB);
    h16* hth = (h16*)((char*)Y + 2 * AFB + (size_t)MTOT * EMB * 2);

    // ---- stage 0: embeddings, taps, degrees, weight conversions ----
    time_emb_k<<<GRID1(NB * 20, 64), 64, 0, stream>>>(t_in, temb);
    node_emb_k<<<GRID1(MTOT * 32, 256), 256, 0, stream>>>(Xt, num_s, 0);
    node_emb_k<<<GRID1(MTOT * 32, 256), 256, 0, stream>>>(Xt, num_t, 1);
    basis_k<<<GRID1(NE, 256), 256, 0, stream>>>(ea_s, tidx_s, tw_s);
    basis_k<<<GRID1(NE, 256), 256, 0, stream>>>(ea_t, tidx_t, tw_t);
    hipMemsetAsync(deg_s, 0, MTOT * 4, stream);
    hipMemsetAsync(deg_t, 0, MTOT * 4, stream);
    deg_count_k<<<GRID1(NE, 256), 256, 0, stream>>>(ei_s, deg_s);
    deg_count_k<<<GRID1(NE, 256), 256, 0, stream>>>(ei_t, deg_t);
    inv_deg_k<<<GRID1(MTOT, 256), 256, 0, stream>>>(deg_s, invd_s);
    inv_deg_k<<<GRID1(MTOT, 256), 256, 0, stream>>>(deg_t, invd_t);
    cvt_xt_k<<<GRID1((long)NB * NSN * NSN, 256), 256, 0, stream>>>(Xt, Xh, XTh);
    cvt_pad_T_k<<<GRID1((long)EMB * 1088, 256), 256, 0, stream>>>(lin0_w, w0hT, 1064, 1088, EMB);
    cvt_pad_T_k<<<GRID1((long)EMB * 192, 256), 256, 0, stream>>>(lin_w, wl0hT, 168, 192, EMB);
    cvt_pad_T_k<<<GRID1((long)EMB * 192, 256), 256, 0, stream>>>(lin_w + 168 * EMB, wl1hT, 168, 192, EMB);
    cvt_pad_T_k<<<GRID1((long)EMB * FIN, 256), 256, 0, stream>>>(final_w, finwhT, FIN, FIN, EMB);

    // ---- 3 layers ----
    for (int l = 0; l < 3; ++l) {
        int Kpad = (l == 0) ? 1088 : 192;
        int W1   = (l == 0) ? ORI_ : EMB;
        const float* prev_s = (l == 0) ? x_s : xls[l - 1];
        const float* prev_t = (l == 0) ? x_t : xlt[l - 1];
        const h16*   whT    = (l == 0) ? w0hT : (l == 1 ? wl0hT : wl1hT);
        const float* lb     = (l == 0) ? lin0_b : lin_b + (l - 1) * EMB;

        // pack linear inputs (concat prev|num|time, zero-padded to Kpad)
        pack_lin_k<<<GRID1((long)MTOT * Kpad, 256), 256, 0, stream>>>(
            prev_s, W1, num_s, temb, Als, Kpad);
        pack_lin_k<<<GRID1((long)MTOT * Kpad, 256), 256, 0, stream>>>(
            prev_t, W1, num_t, temb, Alt, Kpad);

        // xin = A @ w + b
        //   f32 -> xin (diff addend), f16 -> cat left half (conv-A operand),
        //   batched-transposed f16 -> xpT (diff-B operand)
        gemm(stream, Als, Kpad, (long)NSN * Kpad, whT, Kpad, 0,
             MTOT, EMB, Kpad, NSN, lb, nullptr, 0, nullptr, 1.f, 0,
             xin_s, EMB, cat_s, CIN, xpT_s, NSN, (long)EMB * NSN);
        gemm(stream, Alt, Kpad, (long)NSN * Kpad, whT, Kpad, 0,
             MTOT, EMB, Kpad, NSN, lb, nullptr, 0, nullptr, 1.f, 0,
             xin_t, EMB, cat_t, CIN, xpT_t, NSN, (long)EMB * NSN);

        // diff_s = xin_s - Xt @ xp_t  -> cat_s right half (f16)
        gemm(stream, Xh, NSN, (long)NSN * NSN, xpT_t, NSN, (long)EMB * NSN,
             MTOT, EMB, NSN, NSN, nullptr, xin_s, EMB, nullptr, -1.f, 0,
             nullptr, 0, cat_s + EMB, CIN);
        // diff_t = xin_t - Xt^T @ xp_s -> cat_t right half (f16)
        gemm(stream, XTh, NSN, (long)NSN * NSN, xpT_s, NSN, (long)EMB * NSN,
             MTOT, EMB, NSN, NSN, nullptr, xin_t, EMB, nullptr, -1.f, 0,
             nullptr, 0, cat_t + EMB, CIN);

        // conv weights for this layer (transposed f16)
        cvt_wyT_k<<<GRID1((long)KK_ * CIN * EMB, 256), 256, 0, stream>>>(
            conv_w + (long)l * KK_ * CIN * EMB, wyhT);
        cvt_pad_T_k<<<GRID1((long)EMB * CIN, 256), 256, 0, stream>>>(
            conv_root + (long)l * CIN * EMB, roothT, CIN, CIN, EMB);

        // --- spline conv, src side ---
        gemm(stream, cat_s, CIN, 0, wyhT, CIN, 0, MTOT, YW, CIN, MTOT,
             nullptr, nullptr, 0, nullptr, 1.f, 0, Y, YW, nullptr, 0);
        hipMemsetAsync(agg, 0, (long)MTOT * EMB * 4, stream);
        edge_agg_k<<<GRID1((long)NE * EMB, 256), 256, 0, stream>>>(
            ei_s, tidx_s, tw_s, Y, agg);
        gemm(stream, cat_s, CIN, 0, roothT, CIN, 0, MTOT, EMB, CIN, MTOT,
             conv_bias + (long)l * EMB, agg, EMB, invd_s, 1.f, 1,
             xls[l], EMB, nullptr, 0);

        // --- spline conv, trg side ---
        gemm(stream, cat_t, CIN, 0, wyhT, CIN, 0, MTOT, YW, CIN, MTOT,
             nullptr, nullptr, 0, nullptr, 1.f, 0, Y, YW, nullptr, 0);
        hipMemsetAsync(agg, 0, (long)MTOT * EMB * 4, stream);
        edge_agg_k<<<GRID1((long)NE * EMB, 256), 256, 0, stream>>>(
            ei_t, tidx_t, tw_t, Y, agg);
        gemm(stream, cat_t, CIN, 0, roothT, CIN, 0, MTOT, EMB, CIN, MTOT,
             conv_bias + (long)l * EMB, agg, EMB, invd_t, 1.f, 1,
             xlt[l], EMB, nullptr, 0);
    }

    // ---- final projection + similarity + softmax ----
    pack_fin_k<<<GRID1((long)MTOT * FIN, 256), 256, 0, stream>>>(
        x_s, xls[0], xls[1], xls[2], Afs);
    pack_fin_k<<<GRID1((long)MTOT * FIN, 256), 256, 0, stream>>>(
        x_t, xlt[0], xlt[1], xlt[2], Aft);
    gemm(stream, Afs, FIN, 0, finwhT, FIN, 0, MTOT, EMB, FIN, MTOT,
         final_b, nullptr, 0, nullptr, 1.f, 0, nullptr, 0, hsh, EMB);
    gemm(stream, Aft, FIN, 0, finwhT, FIN, 0, MTOT, EMB, FIN, MTOT,
         final_b, nullptr, 0, nullptr, 1.f, 0, nullptr, 0, hth, EMB);
    // sim[b,s,t] = h_s[b,s,:] . h_t[b,t,:]  (B^T == h_t row-major)
    gemm(stream, hsh, EMB, (long)NSN * EMB, hth, EMB, (long)NSN * EMB,
         MTOT, NSN, EMB, NSN, nullptr, nullptr, 0, nullptr, 1.f, 0,
         out, NSN, nullptr, 0);
    softmax512_k<<<dim3(MTOT), dim3(256), 0, stream>>>(out);

    (void)in_sizes; (void)n_in; (void)out_size; (void)ws_size;
}
